// HeteroNGCF_49976239456890
// MI455X (gfx1250) — matrix operand order repaired and checked by reading, not compiled, not run on hardware
//
#include <hip/hip_runtime.h>
#include <hip/hip_bf16.h>

#define D 128
#define TILE_E 32
#define NEG_SLOPE 0.01f
#define LN_EPS 1e-5f

typedef __attribute__((ext_vector_type(16))) __bf16 v16bf;
typedef __attribute__((ext_vector_type(8)))  float  v8f;

static inline int imin_host(int a, int b) { return a < b ? a : b; }

// ---------------------------------------------------------------------------
// Zero the accumulation workspace (must run every launch: atomics accumulate).
// ---------------------------------------------------------------------------
__global__ void zero_ws_kernel(float4* __restrict__ p, long n4) {
  long i = (long)blockIdx.x * blockDim.x + threadIdx.x;
  long stride = (long)gridDim.x * blockDim.x;
  float4 z; z.x = z.y = z.z = z.w = 0.0f;
  for (; i < n4; i += stride) p[i] = z;
}

// ---------------------------------------------------------------------------
// NGCF edge conv: agg[dst] += LeakyReLU((x_src[src] * x_dst[dst]) @ W + b)
// 256 threads = 8 waves. Per 32-edge tile: cooperative gather/product into
// LDS A-fragments (bf16, two M-tiles), wave w computes N-tile w for both
// M-tiles via v_wmma_f32_16x16x32_bf16 (B fragment reused), then fused
// bias+LeakyReLU+atomic scatter with dst indices staged in LDS.
// W staged once per block, pre-swizzled to the B-fragment VGPR layout.
// ---------------------------------------------------------------------------
__launch_bounds__(256)
__global__ void ngcf_edge_conv_kernel(
    const float* __restrict__ xsrc, const float* __restrict__ xdst,
    const float* __restrict__ W,    const float* __restrict__ bias,
    const int* __restrict__ esrc,   const int* __restrict__ edst,
    int E, float* __restrict__ agg)
{
  // [kt(4)][nt(8)][lane(32)][16 bf16] = 32 KB
  __shared__ __align__(32) __bf16 sW[4 * 8 * 32 * 16];
  // [mt(2)][kt(4)][lane(32)][16 bf16] = 8 KB
  __shared__ __align__(32) __bf16 sA[2 * 4 * 32 * 16];
  __shared__ __align__(16) int sDst[TILE_E];

  const int t = threadIdx.x;

  // ---- Stage W (k-major in memory: W[k*D + n]) into B-fragment layout ----
  // B 32x16 bf16 layout used here (same K order as A): lane<16 -> (K = kt*32 + j, N = nt*16 + lane)
  //                      lane>=16 -> (K = kt*32 + 16 + j, N = nt*16 + lane-16)
  for (int f = t; f < 1024; f += 256) {
    int kt   = f >> 8;
    int r    = f & 255;
    int nt   = r >> 5;
    int lane = r & 31;
    int n    = nt * 16 + (lane & 15);
    int kb   = kt * 32 + ((lane & 16) ? 16 : 0);
    __bf16* o = &sW[f * 16];
    #pragma unroll
    for (int j = 0; j < 16; ++j)
      o[j] = (__bf16)W[(kb + j) * D + n];
  }
  __syncthreads();

  const int numTiles = (E + TILE_E - 1) / TILE_E;
  const int wv = t >> 5;   // wave id == N-tile index (0..7)
  const int l  = t & 31;   // lane id
  const int n  = wv * 16 + (l & 15);     // output column owned by this lane
  const float bn = bias[n];
  float* const aggN = agg + n;

  for (int tile = blockIdx.x; tile < numTiles; tile += gridDim.x) {
    const int e0 = tile * TILE_E;

    // ---- Produce A fragments: bf16( x_src[src[e]] * x_dst[dst[e]] ) ----
    // thread t: edge m = t/8 (0..31), 16-K chunk gs = t%8 -> K in [16gs,16gs+16)
    // A 16x32 bf16 layout used here (per M-tile): lane<16 holds (M=lane, K 0..15
    // of the K-tile), lane>=16 holds (M=lane-16, K 16..31).
    {
      int m  = t >> 3;
      int gs = t & 7;
      int e  = e0 + m;
      int mt = m >> 4;
      int ml = m & 15;
      if (gs == 0) sDst[m] = (e < E) ? edst[e] : -1;
      __align__(16) __bf16 h[16];
      if (e < E) {
        long s = esrc[e];
        long d = edst[e];
        const float4* xs = (const float4*)(xsrc + s * D) + 4 * gs;
        const float4* xd = (const float4*)(xdst + d * D) + 4 * gs;
        #pragma unroll
        for (int q = 0; q < 4; ++q) {
          float4 a = xs[q], b = xd[q];
          h[4 * q + 0] = (__bf16)(a.x * b.x);
          h[4 * q + 1] = (__bf16)(a.y * b.y);
          h[4 * q + 2] = (__bf16)(a.z * b.z);
          h[4 * q + 3] = (__bf16)(a.w * b.w);
        }
      } else {
        #pragma unroll
        for (int j = 0; j < 16; ++j) h[j] = (__bf16)0.0f;
      }
      #pragma unroll
      for (int r = 0; r < 2; ++r) {       // two 8-K runs per thread
        int g    = 2 * gs + r;            // 8-run index within the row
        int kt   = g >> 2;                // K-tile (32 K each)
        int sub  = g & 3;
        int lane = ml + ((sub & 2) ? 16 : 0);
        int eb   = (sub & 1) ? 8 : 0;
        *(uint4*)&sA[(((mt * 4 + kt) * 32 + lane) * 16) + eb] =
            *(const uint4*)&h[8 * r];
      }
    }
    __syncthreads();

    // ---- GEMM: wave wv computes two 16(M) x 16(N=wv) tiles over K=128 ----
    v8f c0 = {}, c1 = {};
    #pragma unroll
    for (int kt = 0; kt < 4; ++kt) {
      v16bf b  = *(const v16bf*)&sW[((kt * 8 + wv) * 32 + l) * 16];
      v16bf a0 = *(const v16bf*)&sA[((kt)     * 32 + l) * 16];
      v16bf a1 = *(const v16bf*)&sA[((4 + kt) * 32 + l) * 16];
      c0 = __builtin_amdgcn_wmma_f32_16x16x32_bf16(
               false, a0, false, b, (short)0, c0, false, false);
      c1 = __builtin_amdgcn_wmma_f32_16x16x32_bf16(
               false, a1, false, b, (short)0, c1, false, false);
    }

    // ---- bias + LeakyReLU + scatter-add ----
    // C layout: VGPR i, lane l -> (M = i + 8*(l>=16), N = l&15)
    {
      int mbase = (l & 16) ? 8 : 0;
      int4 dA = *(const int4*)&sDst[mbase];
      int4 dB = *(const int4*)&sDst[mbase + 4];
      int4 dC = *(const int4*)&sDst[16 + mbase];
      int4 dD = *(const int4*)&sDst[16 + mbase + 4];
      int dl[16];
      dl[0] = dA.x;  dl[1] = dA.y;  dl[2] = dA.z;  dl[3] = dA.w;
      dl[4] = dB.x;  dl[5] = dB.y;  dl[6] = dB.z;  dl[7] = dB.w;
      dl[8] = dC.x;  dl[9] = dC.y;  dl[10] = dC.z; dl[11] = dC.w;
      dl[12] = dD.x; dl[13] = dD.y; dl[14] = dD.z; dl[15] = dD.w;
      float cv[16];
      #pragma unroll
      for (int i = 0; i < 8; ++i) { cv[i] = c0[i]; cv[8 + i] = c1[i]; }

      if (e0 + TILE_E <= E) {
        // full tile: branch-free scatter
        #pragma unroll
        for (int i = 0; i < 16; ++i) {
          float v = cv[i] + bn;
          v = (v > 0.0f) ? v : v * NEG_SLOPE;
          atomicAdd(&aggN[(long)dl[i] * D], v);
        }
      } else {
        #pragma unroll
        for (int i = 0; i < 16; ++i) {
          if (dl[i] >= 0) {
            float v = cv[i] + bn;
            v = (v > 0.0f) ? v : v * NEG_SLOPE;
            atomicAdd(&aggN[(long)dl[i] * D], v);
          }
        }
      }
    }
    __syncthreads();  // protect sA/sDst before next iteration's producers
  }
}

// ---------------------------------------------------------------------------
// Per-node LayerNorm + ReLU. One wave32 per node (4 f32 per lane).
// ---------------------------------------------------------------------------
__launch_bounds__(256)
__global__ void ln_relu_kernel(const float* __restrict__ agg,
                               const float* __restrict__ w,
                               const float* __restrict__ b,
                               float* __restrict__ out, int nNodes)
{
  int wave = threadIdx.x >> 5;
  int l    = threadIdx.x & 31;
  int node = blockIdx.x * 8 + wave;
  if (node >= nNodes) return;

  float4 x = ((const float4*)(agg + (long)node * D))[l];

  float s = x.x + x.y + x.z + x.w;
  #pragma unroll
  for (int o = 16; o > 0; o >>= 1) s += __shfl_xor(s, o, 32);
  float mu = s * (1.0f / 128.0f);

  float d0 = x.x - mu, d1 = x.y - mu, d2 = x.z - mu, d3 = x.w - mu;
  float ss = d0 * d0 + d1 * d1 + d2 * d2 + d3 * d3;
  #pragma unroll
  for (int o = 16; o > 0; o >>= 1) ss += __shfl_xor(ss, o, 32);
  float rs = rsqrtf(ss * (1.0f / 128.0f) + LN_EPS);

  float4 wv = ((const float4*)w)[l];
  float4 bv = ((const float4*)b)[l];
  float4 o4;
  o4.x = fmaxf(d0 * rs * wv.x + bv.x, 0.0f);
  o4.y = fmaxf(d1 * rs * wv.y + bv.y, 0.0f);
  o4.z = fmaxf(d2 * rs * wv.z + bv.z, 0.0f);
  o4.w = fmaxf(d3 * rs * wv.w + bv.w, 0.0f);
  ((float4*)(out + (long)node * D))[l] = o4;
}

// ---------------------------------------------------------------------------
extern "C" void kernel_launch(void* const* d_in, const int* in_sizes, int n_in,
                              void* d_out, int out_size, void* d_ws, size_t ws_size,
                              hipStream_t stream) {
  const float* x_user    = (const float*)d_in[0];
  const float* x_item    = (const float*)d_in[1];
  const float* W_ui      = (const float*)d_in[2];
  const float* b_ui      = (const float*)d_in[3];
  const float* W_iu      = (const float*)d_in[4];
  const float* b_iu      = (const float*)d_in[5];
  const float* ln_w_user = (const float*)d_in[6];
  const float* ln_b_user = (const float*)d_in[7];
  const float* ln_w_item = (const float*)d_in[8];
  const float* ln_b_item = (const float*)d_in[9];
  const int* es_ui = (const int*)d_in[10];
  const int* ed_ui = (const int*)d_in[11];
  const int* es_iu = (const int*)d_in[12];
  const int* ed_iu = (const int*)d_in[13];

  const int nUser = in_sizes[0] / D;
  const int nItem = in_sizes[1] / D;
  const int Eui   = in_sizes[10];
  const int Eiu   = in_sizes[12];

  float* aggU = (float*)d_ws;                       // nUser x 128
  float* aggI = aggU + (size_t)nUser * D;           // nItem x 128

  // Zero the accumulators (atomics below accumulate into them).
  long n4 = ((long)nUser + (long)nItem) * (D / 4);
  zero_ws_kernel<<<2048, 256, 0, stream>>>((float4*)d_ws, n4);

  // user->item relation: agg_item += LeakyReLU((x_user[src]*x_item[dst]) @ W_ui + b_ui)
  int tUi = (Eui + TILE_E - 1) / TILE_E;
  ngcf_edge_conv_kernel<<<imin_host(tUi, 4096), 256, 0, stream>>>(
      x_user, x_item, W_ui, b_ui, es_ui, ed_ui, Eui, aggI);

  // item->user relation: agg_user += LeakyReLU((x_item[src]*x_user[dst]) @ W_iu + b_iu)
  int tIu = (Eiu + TILE_E - 1) / TILE_E;
  ngcf_edge_conv_kernel<<<imin_host(tIu, 4096), 256, 0, stream>>>(
      x_item, x_user, W_iu, b_iu, es_iu, ed_iu, Eiu, aggU);

  // LayerNorm(node) + ReLU -> d_out = [out_user | out_item]
  float* outU = (float*)d_out;
  float* outI = outU + (size_t)nUser * D;
  ln_relu_kernel<<<(nUser + 7) / 8, 256, 0, stream>>>(aggU, ln_w_user, ln_b_user, outU, nUser);
  ln_relu_kernel<<<(nItem + 7) / 8, 256, 0, stream>>>(aggI, ln_w_item, ln_b_item, outI, nItem);
}
